// LSTMPhonemeClassifier_42941083025938
// MI455X (gfx1250) — compile-verified, matrix operand order checked by reading, not run on hardware
//
#include <hip/hip_runtime.h>
#include <math.h>

#define SEQ   8192
#define DIM   256
#define HID   1024
#define KCODE 512
#define NCLS  50
#define NWG   256      // persistent workgroups in the scan; each owns 4 hidden units

typedef __attribute__((ext_vector_type(16))) _Float16 v16h;
typedef __attribute__((ext_vector_type(8)))  float    v8f;

__device__ __forceinline__ float sigmoidf_(float x) { return 1.0f / (1.0f + __expf(-x)); }

// K index within a 32-wide chunk for (half, i) per the 16-bit WMMA operand layout:
// i in [0,8): k = half*8 + i ; i in [8,16): k = 16 + half*8 + (i-8)
__device__ __forceinline__ int wmma_k_of(int half, int i)
{
    return (i < 8) ? (half * 8 + i) : (16 + half * 8 + (i - 8));
}

// ---------------------------------------------------------------------------
// Prep 1: codebook -> f16, pre-swizzled into per-lane WMMA B-operand layout.
// cb16[((nt*8 + c)*32 + lane)*16 + i] = cb[nt*16 + (lane&15)][c*32 + k(half,i)]
// 8192 items (nt,c,lane), 16 halves each = 32 contiguous bytes per item.
// ---------------------------------------------------------------------------
__global__ __launch_bounds__(256) void prep_cb_kernel(
    const float* __restrict__ cb, _Float16* __restrict__ cb16)
{
    const int id   = blockIdx.x * 256 + threadIdx.x;   // 0..8191
    const int nt   = id >> 8;
    const int c    = (id >> 5) & 7;
    const int lane = id & 31;
    const int m    = lane & 15;
    const int half = lane >> 4;
    const float* row = cb + (size_t)(nt * 16 + m) * DIM + c * 32;
    _Float16* dst = cb16 + (size_t)id * 16;
#pragma unroll
    for (int i = 0; i < 16; ++i)
        dst[i] = (_Float16)row[wmma_k_of(half, i)];
}

// ---------------------------------------------------------------------------
// Prep 2: x -> f16, pre-swizzled into per-lane WMMA A-operand layout.
// xa16[((tile*8 + c)*32 + lane)*16 + i] = x[tile*16 + (lane&15)][c*32 + k(half,i)]
// 131072 items.
// ---------------------------------------------------------------------------
__global__ __launch_bounds__(256) void prep_x_kernel(
    const float* __restrict__ x, _Float16* __restrict__ xa16)
{
    const int id   = blockIdx.x * 256 + threadIdx.x;   // 0..131071
    const int tile = id >> 8;
    const int c    = (id >> 5) & 7;
    const int lane = id & 31;
    const int m    = lane & 15;
    const int half = lane >> 4;
    const float* row = x + (size_t)(tile * 16 + m) * DIM + c * 32;
    _Float16* dst = xa16 + (size_t)id * 16;
#pragma unroll
    for (int i = 0; i < 16; ++i)
        dst[i] = (_Float16)row[wmma_k_of(half, i)];
}

// ---------------------------------------------------------------------------
// Prep 3: exact f32 squared norms of codebook rows (512) and x rows (8192).
// ---------------------------------------------------------------------------
__global__ __launch_bounds__(256) void prep_norms_kernel(
    const float* __restrict__ cb, const float* __restrict__ x,
    float* __restrict__ cnorm, float* __restrict__ xnorm)
{
    const int t = blockIdx.x * 256 + threadIdx.x;      // 0..8703
    if (t < KCODE) {
        const float* row = cb + (size_t)t * DIM;
        float s = 0.f;
        for (int k = 0; k < DIM; ++k) s += row[k] * row[k];
        cnorm[t] = s;
    } else if (t < KCODE + SEQ) {
        const float* row = x + (size_t)(t - KCODE) * DIM;
        float s = 0.f;
        for (int k = 0; k < DIM; ++k) s += row[k] * row[k];
        xnorm[t - KCODE] = s;
    }
}

// ---------------------------------------------------------------------------
// Kernel 1: VQ argmin. Pure load + WMMA inner loop over pre-swizzled f16 data.
// ---------------------------------------------------------------------------
__global__ __launch_bounds__(256) void vq_argmin_kernel(
    const _Float16* __restrict__ xa16, const _Float16* __restrict__ cb16,
    const float* __restrict__ xnorm,   const float* __restrict__ cnorm,
    int* __restrict__ idx)
{
    const int wave = threadIdx.x >> 5;
    const int lane = threadIdx.x & 31;
    const int m    = lane & 15;
    const int half = lane >> 4;
    const int tile = blockIdx.x * 8 + wave;            // 512 row-tiles total

    const v16h* xa = (const v16h*)xa16;
    const v16h* cbv = (const v16h*)cb16;

    v16h A[8];
#pragma unroll
    for (int c = 0; c < 8; ++c)
        A[c] = xa[(size_t)(tile * 8 + c) * 32 + lane];

    float xn[8];
#pragma unroll
    for (int r = 0; r < 8; ++r)
        xn[r] = xnorm[tile * 16 + r + 8 * half];       // norm of row M = r + 8*half

    float minv[8]; int mini[8];
#pragma unroll
    for (int r = 0; r < 8; ++r) { minv[r] = 3.0e38f; mini[r] = 0; }

    for (int nt = 0; nt < 32; ++nt) {
        // prefetch next code tile's slice for this lane (global_prefetch_b8)
        if (nt < 31)
            __builtin_prefetch(&cbv[(size_t)((nt + 1) * 8) * 32 + lane], 0, 1);

        v8f acc = {};
#pragma unroll
        for (int c = 0; c < 8; ++c) {
            v16h b = cbv[(size_t)(nt * 8 + c) * 32 + lane];
            acc = __builtin_amdgcn_wmma_f32_16x16x32_f16(
                      false, A[c], false, b, (short)0, acc, false, false);
        }
        const float cn   = cnorm[nt * 16 + m];         // |code_N|^2 for N = lane&15
        const int   nidx = nt * 16 + m;
#pragma unroll
        for (int r = 0; r < 8; ++r) {
            float d2 = xn[r] - 2.0f * acc[r] + cn;
            if (d2 < minv[r]) { minv[r] = d2; mini[r] = nidx; }
        }
    }

    // Reduce each C slot over its 16 lanes (xor 1,2,4,8 stays within each half).
#pragma unroll
    for (int r = 0; r < 8; ++r) {
        for (int off = 8; off >= 1; off >>= 1) {
            float ov = __shfl_xor(minv[r], off, 32);
            int   oi = __shfl_xor(mini[r], off, 32);
            if (ov < minv[r]) { minv[r] = ov; mini[r] = oi; }
        }
        if (m == 0) idx[tile * 16 + r + 8 * half] = mini[r];
    }
}

// ---------------------------------------------------------------------------
// Device-wide sense-reversing barrier (self-cleaning: cnt returns to 0 every
// round, gen strictly increases; both memset to 0 at launch).
// ---------------------------------------------------------------------------
__device__ __forceinline__ void grid_barrier(int* cnt, int* gen, int n)
{
    __syncthreads();
    if (threadIdx.x == 0) {
        __threadfence();
        int myg = atomicAdd(gen, 0);
        int old = atomicAdd(cnt, 1);
        if (old == n - 1) {
            atomicExch(cnt, 0);
            __threadfence();
            atomicAdd(gen, 1);
        } else {
            while (atomicAdd(gen, 0) == myg) { __builtin_amdgcn_s_sleep(2); }
        }
    }
    __syncthreads();
}

// ---------------------------------------------------------------------------
// Kernel 2: persistent LSTM scan. WG g owns hidden units u in [4g,4g+4).
// Its 16 gate rows (q*1024 + 4g + u) of W_hh live in LDS (exactly 64 KB).
// One grid barrier per step; h ping-pongs between two global buffers.
// ---------------------------------------------------------------------------
__global__ __launch_bounds__(256) void lstm_scan_kernel(
    const int*   __restrict__ idx,
    const float* __restrict__ Wih,  const float* __restrict__ Whh,
    const float* __restrict__ bih,  const float* __restrict__ bhh,
    const float* __restrict__ h0,   const float* __restrict__ c0,
    float* hbuf,            // 2 * HID floats (ping-pong)
    float* gpre_all,        // NWG * 16 floats (gate preactivation staging)
    int* bar_cnt, int* bar_gen)
{
    __shared__ float wlds[16 * HID];       // 64 KB, the whole W_hh slice for this WG

    const int g   = blockIdx.x;
    const int tid = threadIdx.x;
    const int rr  = tid >> 4;              // row-group 0..15  (q = rr>>2, u = rr&3)
    const int sub = tid & 15;              // 16 threads per row

    // Stage W_hh rows into LDS.
    for (int r = 0; r < 16; ++r) {
        const int grow = ((r >> 2) << 10) + g * 4 + (r & 3);  // q*1024 + 4g + u
        for (int k = tid; k < HID; k += 256)
            wlds[r * HID + k] = Whh[(size_t)grow * HID + k];
    }

    // Init h (global, our slice) and c (registers of threads 0..3).
    float creg = 0.f;
    if (tid < 4) {
        volatile float* hw = (volatile float*)hbuf;
        hw[g * 4 + tid] = h0[g * 4 + tid];
        creg = c0[g * 4 + tid];
        __threadfence();
    }
    grid_barrier(bar_cnt, bar_gen, NWG);   // all of h[0..1023] visible

    volatile float* gp = (volatile float*)(gpre_all + (size_t)g * 16);
    const int grow_me = ((rr >> 2) << 10) + g * 4 + (rr & 3);
    const float bias_me = bhh[grow_me] + bih[grow_me];
    const float* wrow = &wlds[rr * HID + sub * 64];

    for (int t = 0; t < SEQ; ++t) {
        volatile const float* hv = (volatile const float*)(hbuf + (t & 1) * HID);

        float acc = 0.f;
#pragma unroll 8
        for (int k = 0; k < 64; ++k)
            acc += wrow[k] * hv[sub * 64 + k];
        // reduce over the 16 lanes sharing this row (stays within wave halves)
        for (int off = 8; off >= 1; off >>= 1)
            acc += __shfl_xor(acc, off, 32);

        if (sub == 0) {
            // gather x_proj element: W_ih[grow, idx[t]] + biases
            float pre = acc + bias_me + Wih[(size_t)grow_me * KCODE + idx[t]];
            gp[rr] = pre;
            __threadfence_block();
        }
        __syncthreads();

        if (tid < 4) {
            const int u = tid;
            float iv = sigmoidf_(gp[0 * 4 + u]);
            float fv = sigmoidf_(gp[1 * 4 + u]);
            float gv = tanhf    (gp[2 * 4 + u]);
            float ov = sigmoidf_(gp[3 * 4 + u]);
            creg = fv * creg + iv * gv;
            volatile float* hw = (volatile float*)(hbuf + ((t + 1) & 1) * HID);
            hw[g * 4 + u] = ov * tanhf(creg);
            __threadfence();
        }
        grid_barrier(bar_cnt, bar_gen, NWG);
    }
}

// ---------------------------------------------------------------------------
// Kernel 3: logits = h_final @ W_out.T + b_out ; log_softmax. One block.
// Final h is in buffer (SEQ & 1) == 0.
// ---------------------------------------------------------------------------
__global__ __launch_bounds__(1024) void head_kernel(
    const float* __restrict__ hfin, const float* __restrict__ Wout,
    const float* __restrict__ bout, float* __restrict__ out)
{
    __shared__ float logits[64];
    const int tid  = threadIdx.x;
    const int wave = tid >> 5;
    const int lane = tid & 31;

    for (int c = wave; c < NCLS; c += 32) {
        float acc = 0.f;
        for (int k = lane; k < HID; k += 32)
            acc += Wout[(size_t)c * HID + k] * hfin[k];
        for (int off = 16; off >= 1; off >>= 1)
            acc += __shfl_xor(acc, off, 32);
        if (lane == 0) logits[c] = acc + bout[c];
    }
    __syncthreads();

    if (tid == 0) {
        float mx = -3.0e38f;
        for (int c = 0; c < NCLS; ++c) mx = fmaxf(mx, logits[c]);
        float s = 0.f;
        for (int c = 0; c < NCLS; ++c) s += __expf(logits[c] - mx);
        float ls = __logf(s);
        for (int c = 0; c < NCLS; ++c) out[c] = logits[c] - mx - ls;
    }
}

// ---------------------------------------------------------------------------
extern "C" void kernel_launch(void* const* d_in, const int* in_sizes, int n_in,
                              void* d_out, int out_size, void* d_ws, size_t ws_size,
                              hipStream_t stream)
{
    (void)in_sizes; (void)n_in; (void)out_size; (void)ws_size;

    const float* x    = (const float*)d_in[0];
    const float* h0   = (const float*)d_in[1];
    const float* c0   = (const float*)d_in[2];
    const float* cb   = (const float*)d_in[3];
    const float* Wih  = (const float*)d_in[4];
    const float* Whh  = (const float*)d_in[5];
    const float* bih  = (const float*)d_in[6];
    const float* bhh  = (const float*)d_in[7];
    const float* Wout = (const float*)d_in[8];
    const float* bout = (const float*)d_in[9];
    float* out = (float*)d_out;

    char* ws = (char*)d_ws;
    int*      idx   = (int*)     (ws + 0);        // SEQ ints            (  32768 B)
    float*    hbuf  = (float*)   (ws + 32768);    // 2*HID floats        (   8192 B)
    float*    gpre  = (float*)   (ws + 40960);    // NWG*16 floats       (  16384 B)
    int*      bar   = (int*)     (ws + 57344);    // cnt, gen            (      8 B)
    float*    cnorm = (float*)   (ws + 65536);    // KCODE floats        (   2048 B)
    float*    xnorm = (float*)   (ws + 67584);    // SEQ floats          (  32768 B)
    _Float16* cb16  = (_Float16*)(ws + 100352);   // KCODE*DIM halves    ( 262144 B)
    _Float16* xa16  = (_Float16*)(ws + 362496);   // SEQ*DIM halves      (4194304 B)

    hipMemsetAsync(bar, 0, 2 * sizeof(int), stream);

    prep_cb_kernel   <<<  32, 256, 0, stream>>>(cb, cb16);
    prep_x_kernel    <<< 512, 256, 0, stream>>>(x, xa16);
    prep_norms_kernel<<<  34, 256, 0, stream>>>(cb, x, cnorm, xnorm);
    vq_argmin_kernel <<<  64, 256, 0, stream>>>(xa16, cb16, xnorm, cnorm, idx);
    lstm_scan_kernel <<< NWG, 256, 0, stream>>>(idx, Wih, Whh, bih, bhh,
                                                h0, c0, hbuf, gpre, bar, bar + 1);
    head_kernel      <<<   1, 1024, 0, stream>>>(hbuf, Wout, bout, out);
}
